// attn_block_22763326669326
// MI455X (gfx1250) — matrix operand order repaired and checked by reading, not compiled, not run on hardware
//
#include <hip/hip_runtime.h>

// ---------------------------------------------------------------------------
// MI455X (gfx1250) flash-attention implementation of the NHWC attn block.
//
// Shapes: B=8, H=W=64 -> N=4096 tokens, C=64, 32 groups (2 ch/group).
// Precision: f16 WMMA (v_wmma_f32_16x16x32_f16) with f32 accumulation,
// softmax in fp32 (exp2 domain, scale 0.125*log2e folded into q).
// Flash formulation: never materializes the 4096x4096 score matrix
// (~1 GB HBM round-trip avoided; kernel is WMMA-bound, not HBM-bound).
// Round 2: K/V chunks staged with CDNA5 GLOBAL_LOAD_ASYNC_TO_LDS_B128
// (ASYNCcnt-tracked) and double-buffered in LDS so staging overlaps WMMA.
// ---------------------------------------------------------------------------

typedef __attribute__((ext_vector_type(16))) _Float16 v16h;
typedef __attribute__((ext_vector_type(8)))  float    v8f;
typedef __attribute__((ext_vector_type(4)))  unsigned int u32x4;

union Frag16 {            // one WMMA 16-bit operand fragment (8 VGPRs/lane)
    v16h     h;
    _Float16 e[16];
    u32x4    q[2];
};
union Pack8 {             // 8 f16 = 16B vector store
    _Float16 e[8];
    u32x4    v;
};

__device__ __forceinline__ v8f wmma_f16(v16h a, v16h b, v8f c) {
    // D = A(16x32 f16) * B(32x16 f16) + C(16x16 f32)
    return __builtin_amdgcn_wmma_f32_16x16x32_f16(
        /*neg_a=*/false, a, /*neg_b=*/false, b,
        /*c_mod=*/(short)0, c, /*reuse_a=*/false, /*reuse_b=*/false);
}

// CDNA5 async global->LDS copy, 16 bytes per lane (GV addressing mode).
// %0 = VGPR holding LDS byte address, %1 = VGPR pair holding 64-bit vaddr.
__device__ __forceinline__ void async_copy_b128(const void* lds_dst,
                                                const void* gsrc) {
    unsigned           ldsoff = (unsigned)(uintptr_t)lds_dst;   // low 32b = LDS addr
    unsigned long long gaddr  = (unsigned long long)(uintptr_t)gsrc;
    asm volatile("global_load_async_to_lds_b128 %0, %1, off"
                 :: "v"(ldsoff), "v"(gaddr)
                 : "memory");
}
__device__ __forceinline__ void wait_asynccnt0() {
    asm volatile("s_wait_asynccnt 0x0" ::: "memory");
}

#define QSCALE (0.125f * 1.44269504088896340736f)   /* C^-0.5 * log2(e) */

// ---------------------------------------------------------------------------
// Kernel 1: group-norm statistics. grid=(32 groups, 8 batches), 256 thr.
// Each (b,g) reduces 4096 spatial * 2 channels.
// ---------------------------------------------------------------------------
__global__ __launch_bounds__(256)
void gn_stats_kernel(const float* __restrict__ x,
                     float* __restrict__ meanp, float* __restrict__ rstdp) {
    const int g = blockIdx.x, b = blockIdx.y, tid = threadIdx.x;
    const float* xb = x + (size_t)b * 4096 * 64 + g * 2;
    float s = 0.f, s2 = 0.f;
    for (int i = tid; i < 4096; i += 256) {
        float a = xb[(size_t)i * 64 + 0];
        float c = xb[(size_t)i * 64 + 1];
        s += a + c;
        s2 += a * a + c * c;
    }
    __shared__ float rs[256], rq[256];
    rs[tid] = s; rq[tid] = s2;
    __syncthreads();
    for (int off = 128; off > 0; off >>= 1) {
        if (tid < off) { rs[tid] += rs[tid + off]; rq[tid] += rq[tid + off]; }
        __syncthreads();
    }
    if (tid == 0) {
        float mean = rs[0] * (1.0f / 8192.0f);
        float var  = rq[0] * (1.0f / 8192.0f) - mean * mean;
        meanp[b * 32 + g] = mean;
        rstdp[b * 32 + g] = rsqrtf(var + 1e-6f);
    }
}

// ---------------------------------------------------------------------------
// Kernel 2: convert Wq/Wk/Wv/Wp (fp32 [in][out]) -> f16 transposed [out][in]
// (B-fragment wants 16 contiguous input-channel values per output column).
// ---------------------------------------------------------------------------
__global__ __launch_bounds__(256)
void wprep_kernel(const float* __restrict__ Wq, const float* __restrict__ Wk,
                  const float* __restrict__ Wv, const float* __restrict__ Wp,
                  _Float16* __restrict__ WT) {
    const float* Ws[4] = {Wq, Wk, Wv, Wp};
    for (int idx = threadIdx.x; idx < 4 * 4096; idx += 256) {
        int mat = idx >> 12;
        int r   = idx & 4095;
        int o   = r >> 6;     // output channel
        int i   = r & 63;     // input channel
        WT[idx] = (_Float16)Ws[mat][i * 64 + o];
    }
}

// ---------------------------------------------------------------------------
// Kernel 3: fused groupnorm-normalize + q/k/v projections via WMMA.
// grid=(64 token-tiles, 8 batches), 128 thr = 4 waves, 16 tokens/wave.
// q stored f16 [b][t][c] pre-scaled by QSCALE, k stored f16 [b][t][c],
// v stored f16 transposed [b][c][t].
// ---------------------------------------------------------------------------
__global__ __launch_bounds__(128)
void qkv_proj_kernel(const float* __restrict__ x,
                     const float* __restrict__ gamma, const float* __restrict__ beta,
                     const float* __restrict__ meanp, const float* __restrict__ rstdp,
                     const _Float16* __restrict__ WT,
                     const float* __restrict__ bq, const float* __restrict__ bk,
                     const float* __restrict__ bv,
                     _Float16* __restrict__ qf, _Float16* __restrict__ kf,
                     _Float16* __restrict__ vTf) {
    const int b     = blockIdx.y;
    const int tbase = blockIdx.x * 64;
    const int tid   = threadIdx.x;
    const int wave  = tid >> 5;
    const int lane  = tid & 31;
    const int lh    = lane & 15;
    const int hi    = lane >> 4;      // lane half
    const int koff  = hi * 8;         // unused; fragments use K base hi * 16

    __shared__ float smean[32], srstd[32], sg[64], sb[64];
    if (tid < 32) { smean[tid] = meanp[b * 32 + tid]; srstd[tid] = rstdp[b * 32 + tid]; }
    if (tid < 64) { sg[tid] = gamma[tid]; sb[tid] = beta[tid]; }
    __syncthreads();

    const int tok = tbase + wave * 16 + lh;          // this lane's A-matrix row
    const float* xp = x + ((size_t)b * 4096 + tok) * 64;

    // Build normalized-h A fragments (16 tokens x 64 ch = 2 chunks of K=32).
    Frag16 a0, a1;
#pragma unroll
    for (int e = 0; e < 8; e++) {
        int c0 = hi * 16 + e, c1 = hi * 16 + 8 + e, c2 = 32 + hi * 16 + e, c3 = 32 + hi * 16 + 8 + e;
        a0.e[e]     = (_Float16)((xp[c0] - smean[c0 >> 1]) * srstd[c0 >> 1] * sg[c0] + sb[c0]);
        a0.e[8 + e] = (_Float16)((xp[c1] - smean[c1 >> 1]) * srstd[c1 >> 1] * sg[c1] + sb[c1]);
        a1.e[e]     = (_Float16)((xp[c2] - smean[c2 >> 1]) * srstd[c2 >> 1] * sg[c2] + sb[c2]);
        a1.e[8 + e] = (_Float16)((xp[c3] - smean[c3 >> 1]) * srstd[c3 >> 1] * sg[c3] + sb[c3]);
    }

    const _Float16* WTs[3]    = {WT, WT + 4096, WT + 2 * 4096};
    const float*    biases[3] = {bq, bk, bv};

#pragma unroll
    for (int pj = 0; pj < 3; pj++) {
        const _Float16* Wt   = WTs[pj];
        const float*    bias = biases[pj];
#pragma unroll
        for (int ct = 0; ct < 4; ct++) {             // 4 output-channel tiles
            Frag16 b0, b1;
            const _Float16* wp = Wt + (ct * 16 + lh) * 64 + hi * 16;
            b0.q[0] = *(const u32x4*)(wp);       b0.q[1] = *(const u32x4*)(wp + 8);
            b1.q[0] = *(const u32x4*)(wp + 32);  b1.q[1] = *(const u32x4*)(wp + 40);

            v8f acc;
            float bb = bias[ct * 16 + lh];
#pragma unroll
            for (int r = 0; r < 8; r++) acc[r] = bb;
            acc = wmma_f16(a0.h, b0.h, acc);
            acc = wmma_f16(a1.h, b1.h, acc);

            if (pj == 0) {
#pragma unroll
                for (int r = 0; r < 8; r++) acc[r] *= QSCALE;
            }
            if (pj < 2) {
                _Float16* dst = (pj == 0) ? qf : kf;
#pragma unroll
                for (int r = 0; r < 8; r++) {
                    int tr = tbase + wave * 16 + hi * 8 + r;
                    dst[((size_t)b * 4096 + tr) * 64 + ct * 16 + lh] = (_Float16)acc[r];
                }
            } else {
                Pack8 pk;                             // vT[ch][token]: 8 tokens contiguous
#pragma unroll
                for (int r = 0; r < 8; r++) pk.e[r] = (_Float16)acc[r];
                *(u32x4*)(vTf + ((size_t)b * 64 + ct * 16 + lh) * 4096
                              + tbase + wave * 16 + hi * 8) = pk.v;
            }
        }
    }
}

// ---------------------------------------------------------------------------
// Kernel 4: flash attention + output projection + residual.
// grid=(64 query-tiles, 8 batches), 128 thr = 4 waves, 16 query rows/wave.
// K/V 64-key chunks double-buffered in LDS via async global->LDS copies.
// ---------------------------------------------------------------------------
__global__ __launch_bounds__(128)
void flash_attn_kernel(const _Float16* __restrict__ qf,
                       const _Float16* __restrict__ kf,
                       const _Float16* __restrict__ vTf,
                       const _Float16* __restrict__ WpT,
                       const float* __restrict__ bp,
                       const float* __restrict__ x,
                       float* __restrict__ out) {
    const int b     = blockIdx.y;
    const int qbase = blockIdx.x * 64;
    const int tid   = threadIdx.x;
    const int wave  = tid >> 5;
    const int lane  = tid & 31;
    const int lh    = lane & 15;
    const int hi    = lane >> 4;
    const int koff  = hi * 8;

    __shared__ _Float16 kbuf[2][64 * 64];    // [key][ch]   2 x 8 KB
    __shared__ _Float16 vbuf[2][64 * 64];    // [ch][key]   2 x 8 KB
    __shared__ _Float16 pbuf[4][16 * 32];    // per-wave P  4 KB
    __shared__ _Float16 obuf[4][16 * 64];    // per-wave O  8 KB

    const _Float16* kB  = kf  + (size_t)b * 4096 * 64;   // chunk kc at +kc*4096 elems
    const _Float16* vTB = vTf + (size_t)b * 64 * 4096;

    // Resident Q fragments (16 rows x 64 ch, pre-scaled).
    const int qtok = qbase + wave * 16 + lh;
    const _Float16* qp = qf + ((size_t)b * 4096 + qtok) * 64;
    Frag16 aq0, aq1;
    aq0.q[0] = *(const u32x4*)(qp + hi * 16);
    aq0.q[1] = *(const u32x4*)(qp + hi * 16 + 8);
    aq1.q[0] = *(const u32x4*)(qp + 32 + hi * 16);
    aq1.q[1] = *(const u32x4*)(qp + 32 + hi * 16 + 8);

    v8f oacc[4];
    float mrow[8], lrow[8];
#pragma unroll
    for (int r = 0; r < 8; r++) {
        mrow[r] = -3.0e38f; lrow[r] = 0.f;
        oacc[0][r] = 0.f; oacc[1][r] = 0.f; oacc[2][r] = 0.f; oacc[3][r] = 0.f;
    }

    // Async-stage one 64-key chunk (8 KB K + 8 KB V): 8 x b128 per thread.
    auto stage_chunk = [&](int kc, int buf) {
        const _Float16* ksrc = kB + (size_t)kc * 64 * 64;      // contiguous 8 KB
#pragma unroll
        for (int j = 0; j < 4; j++) {
            int off = tid * 8 + j * 1024;                       // f16 elements
            async_copy_b128(&kbuf[buf][off], ksrc + off);
        }
#pragma unroll
        for (int j = 0; j < 4; j++) {
            int s = tid + j * 128, ch = s >> 3, sg = s & 7;     // 64 rows x 8 segs
            async_copy_b128(&vbuf[buf][ch * 64 + sg * 8],
                            vTB + (size_t)ch * 4096 + kc * 64 + sg * 8);
        }
    };

    stage_chunk(0, 0);          // prologue fill of buffer 0
    wait_asynccnt0();
    __syncthreads();

    for (int kc = 0; kc < 64; kc++) {                 // 64-key chunks
        const int cur = kc & 1;
        if (kc + 1 < 64)
            stage_chunk(kc + 1, cur ^ 1);             // overlap with compute below
        const _Float16* kbufc = kbuf[cur];
        const _Float16* vbufc = vbuf[cur];

#pragma unroll
        for (int ks = 0; ks < 2; ks++) {              // 32-key steps
            // --- S = Q * K^T, two 16-key subtiles ---
            v8f s0, s1;
#pragma unroll
            for (int r = 0; r < 8; r++) { s0[r] = 0.f; s1[r] = 0.f; }
            {
                const _Float16* kp0 = kbufc + (ks * 32 + lh) * 64 + hi * 16;
                const _Float16* kp1 = kbufc + (ks * 32 + 16 + lh) * 64 + hi * 16;
                Frag16 bk0, bk1;
                bk0.q[0] = *(const u32x4*)(kp0);      bk0.q[1] = *(const u32x4*)(kp0 + 8);
                bk1.q[0] = *(const u32x4*)(kp0 + 32); bk1.q[1] = *(const u32x4*)(kp0 + 40);
                s0 = wmma_f16(aq0.h, bk0.h, s0);
                s0 = wmma_f16(aq1.h, bk1.h, s0);
                bk0.q[0] = *(const u32x4*)(kp1);      bk0.q[1] = *(const u32x4*)(kp1 + 8);
                bk1.q[0] = *(const u32x4*)(kp1 + 32); bk1.q[1] = *(const u32x4*)(kp1 + 40);
                s1 = wmma_f16(aq0.h, bk0.h, s1);
                s1 = wmma_f16(aq1.h, bk1.h, s1);
            }

            // --- online softmax (exp2 domain) ---
            float cf[8];
#pragma unroll
            for (int r = 0; r < 8; r++) {
                float cur_m = fmaxf(s0[r], s1[r]);
#pragma unroll
                for (int d = 1; d < 16; d <<= 1)      // 16-lane row butterfly
                    cur_m = fmaxf(cur_m, __shfl_xor(cur_m, d, 32));
                float mnew = fmaxf(mrow[r], cur_m);
                cf[r] = exp2f(mrow[r] - mnew);
                mrow[r] = mnew;
                s0[r] = exp2f(s0[r] - mnew);
                s1[r] = exp2f(s1[r] - mnew);
                float rsum = s0[r] + s1[r];
#pragma unroll
                for (int d = 1; d < 16; d <<= 1)
                    rsum += __shfl_xor(rsum, d, 32);
                lrow[r] = lrow[r] * cf[r] + rsum;
                oacc[0][r] *= cf[r]; oacc[1][r] *= cf[r];
                oacc[2][r] *= cf[r]; oacc[3][r] *= cf[r];
            }

            // --- P: D-layout -> A-layout through wave-private LDS ---
            _Float16* pb = pbuf[wave];
#pragma unroll
            for (int r = 0; r < 8; r++) {
                int row = r + hi * 8;
                pb[row * 32 + lh]      = (_Float16)s0[r];
                pb[row * 32 + 16 + lh] = (_Float16)s1[r];
            }
            Frag16 ap;                                 // same-wave LDS is in-order
            const _Float16* pr = pb + lh * 32 + hi * 16;
            ap.q[0] = *(const u32x4*)(pr);
            ap.q[1] = *(const u32x4*)(pr + 8);

            // --- O += P * V (4 channel tiles) ---
#pragma unroll
            for (int ct = 0; ct < 4; ct++) {
                Frag16 bvf;
                const _Float16* vp = vbufc + (ct * 16 + lh) * 64 + ks * 32 + hi * 16;
                bvf.q[0] = *(const u32x4*)(vp);
                bvf.q[1] = *(const u32x4*)(vp + 8);
                oacc[ct] = wmma_f16(ap.h, bvf.h, oacc[ct]);
            }
        }

        // Drain this wave's async loads for chunk kc+1, then block-wide
        // barrier: (a) makes buf[cur^1] contents visible to all waves,
        // (b) guarantees all reads of buf[cur] finished before iteration
        // kc+1 issues async writes into it.
        wait_asynccnt0();
        __syncthreads();
    }

    // --- epilogue: normalize, project with Wp, add bias + residual ---
    float inv[8];
#pragma unroll
    for (int r = 0; r < 8; r++) inv[r] = 1.0f / lrow[r];

    _Float16* ob = obuf[wave];
#pragma unroll
    for (int ct = 0; ct < 4; ct++)
#pragma unroll
        for (int r = 0; r < 8; r++)
            ob[(r + hi * 8) * 64 + ct * 16 + lh] = (_Float16)(oacc[ct][r] * inv[r]);

    Frag16 ao0, ao1;
    const _Float16* orp = ob + lh * 64;
    ao0.q[0] = *(const u32x4*)(orp + hi * 16);
    ao0.q[1] = *(const u32x4*)(orp + hi * 16 + 8);
    ao1.q[0] = *(const u32x4*)(orp + 32 + hi * 16);
    ao1.q[1] = *(const u32x4*)(orp + 32 + hi * 16 + 8);

#pragma unroll
    for (int ct = 0; ct < 4; ct++) {
        Frag16 bw0, bw1;
        const _Float16* wp = WpT + (ct * 16 + lh) * 64 + hi * 16;
        bw0.q[0] = *(const u32x4*)(wp);      bw0.q[1] = *(const u32x4*)(wp + 8);
        bw1.q[0] = *(const u32x4*)(wp + 32); bw1.q[1] = *(const u32x4*)(wp + 40);

        v8f f;
        float bias = bp[ct * 16 + lh];
#pragma unroll
        for (int r = 0; r < 8; r++) {
            int tr = qbase + wave * 16 + hi * 8 + r;
            f[r] = bias + x[((size_t)b * 4096 + tr) * 64 + ct * 16 + lh];
        }
        f = wmma_f16(ao0.h, bw0.h, f);
        f = wmma_f16(ao1.h, bw1.h, f);
#pragma unroll
        for (int r = 0; r < 8; r++) {
            int tr = qbase + wave * 16 + hi * 8 + r;
            out[((size_t)b * 4096 + tr) * 64 + ct * 16 + lh] = f[r];
        }
    }
}

// ---------------------------------------------------------------------------
// Launch
// ---------------------------------------------------------------------------
extern "C" void kernel_launch(void* const* d_in, const int* in_sizes, int n_in,
                              void* d_out, int out_size, void* d_ws, size_t ws_size,
                              hipStream_t stream) {
    (void)in_sizes; (void)n_in; (void)out_size; (void)ws_size;
    const float* x     = (const float*)d_in[0];
    const float* gamma = (const float*)d_in[1];
    const float* beta  = (const float*)d_in[2];
    const float* Wq    = (const float*)d_in[3];
    const float* bq    = (const float*)d_in[4];
    const float* Wk    = (const float*)d_in[5];
    const float* bk    = (const float*)d_in[6];
    const float* Wv    = (const float*)d_in[7];
    const float* bv    = (const float*)d_in[8];
    const float* Wp    = (const float*)d_in[9];
    const float* bp    = (const float*)d_in[10];
    float* out = (float*)d_out;

    char* ws = (char*)d_ws;
    float*     meanp = (float*)ws;                    // 256 f32
    float*     rstdp = meanp + 256;                   // 256 f32
    _Float16*  WT    = (_Float16*)(ws + 4096);        // 4*64*64 f16 = 32 KB
    _Float16*  qf    = (_Float16*)(ws + 4096 + 32768);
    const size_t NTOT = (size_t)8 * 4096 * 64;        // 2M f16 each
    _Float16*  kfb   = qf + NTOT;
    _Float16*  vTf   = kfb + NTOT;

    gn_stats_kernel<<<dim3(32, 8), 256, 0, stream>>>(x, meanp, rstdp);
    wprep_kernel<<<dim3(1), 256, 0, stream>>>(Wq, Wk, Wv, Wp, WT);
    qkv_proj_kernel<<<dim3(64, 8), 128, 0, stream>>>(x, gamma, beta, meanp, rstdp,
                                                     WT, bq, bk, bv, qf, kfb, vTf);
    flash_attn_kernel<<<dim3(64, 8), 128, 0, stream>>>(qf, kfb, vTf, WT + 3 * 4096,
                                                       bp, x, out);
}
